// Model_51032801411125
// MI455X (gfx1250) — compile-verified, matrix operand order checked by reading, not run on hardware
//
#include <hip/hip_runtime.h>
#include <hip/hip_bf16.h>
#include <stdint.h>

#define N_USERN 100000
#define N_ITEMN 50000
#define NEDGE   600000
#define D_INN   128
#define HN      256

typedef __attribute__((ext_vector_type(16))) __bf16 v16bf;
typedef __attribute__((ext_vector_type(8)))  float  v8f;

static __device__ __forceinline__ unsigned short f2bf(float f) {
  union { float f; unsigned u; } x; x.f = f;
  unsigned u = x.u;
  unsigned r = u + 0x7FFFu + ((u >> 16) & 1u);   // round-to-nearest-even
  return (unsigned short)(r >> 16);
}

// ---- weight convert + transpose: W[K x N] f32 -> Wt[N x K] bf16 ----
__global__ void wconv_kernel(const float* __restrict__ W,
                             unsigned short* __restrict__ Wt, int K, int N) {
  int t = blockIdx.x * blockDim.x + threadIdx.x;
  if (t >= K * N) return;
  int n = t / K, k = t % K;
  Wt[t] = f2bf(W[k * N + n]);
}

// ---- activation convert: Y_bf16 = X_f32 * (scale ? scale[row] : 1), 4 elems/thread ----
__global__ void cvt_kernel(const float* __restrict__ X, const float* __restrict__ scale,
                           unsigned short* __restrict__ Y, int M, int K) {
  long long t = (long long)blockIdx.x * blockDim.x + threadIdx.x;
  long long total = (long long)M * K / 4;
  if (t >= total) return;
  long long i4 = t * 4;
  int row = (int)(i4 / K);
  float rs = scale ? scale[row] : 1.0f;
  float4 v = *(const float4*)(X + i4);
  ushort4 o;
  o.x = f2bf(v.x * rs); o.y = f2bf(v.y * rs);
  o.z = f2bf(v.z * rs); o.w = f2bf(v.w * rs);
  *(ushort4*)(Y + i4) = o;
}

// ---- per-destination edge count ----
__global__ void count_kernel(const int* __restrict__ dst, float* __restrict__ cnt, int E) {
  int t = blockIdx.x * blockDim.x + threadIdx.x;
  if (t >= E) return;
  __hip_atomic_fetch_add(&cnt[dst[t]], 1.0f, __ATOMIC_RELAXED, __HIP_MEMORY_SCOPE_AGENT);
}

// ---- cnt -> 1/max(cnt,1) in place ----
__global__ void rinv_kernel(float* __restrict__ c, int n) {
  int t = blockIdx.x * blockDim.x + threadIdx.x;
  if (t >= n) return;
  c[t] = 1.0f / fmaxf(c[t], 1.0f);
}

// ---- scatter-sum: agg[dst[e], :] += X[src[e], :], 4 floats per thread ----
__global__ void scatter_kernel(const float* __restrict__ X, const int* __restrict__ src,
                               const int* __restrict__ dst, float* __restrict__ agg,
                               int E, int D) {
  int perEdge = D >> 2;
  long long t = (long long)blockIdx.x * blockDim.x + threadIdx.x;
  if (t >= (long long)E * perEdge) return;
  int e  = (int)(t / perEdge);
  int c4 = ((int)(t % perEdge)) << 2;
  int s = src[e], d = dst[e];
  float4 v = *(const float4*)(X + (size_t)s * D + c4);
  float* p = agg + (size_t)d * D + c4;
  __hip_atomic_fetch_add(p + 0, v.x, __ATOMIC_RELAXED, __HIP_MEMORY_SCOPE_AGENT);
  __hip_atomic_fetch_add(p + 1, v.y, __ATOMIC_RELAXED, __HIP_MEMORY_SCOPE_AGENT);
  __hip_atomic_fetch_add(p + 2, v.z, __ATOMIC_RELAXED, __HIP_MEMORY_SCOPE_AGENT);
  __hip_atomic_fetch_add(p + 3, v.w, __ATOMIC_RELAXED, __HIP_MEMORY_SCOPE_AGENT);
}

// ---- bf16 fragment load: two contiguous 16B chunks per lane ----
static __device__ __forceinline__ v16bf load_frag(const unsigned short* p, int kb, int koff) {
  union { v16bf v; uint4 q[2]; } u;
  u.q[0] = *(const uint4*)(p + kb + koff);
  u.q[1] = *(const uint4*)(p + kb + 16 + koff);
  return u.v;
}

// ---- fused SAGE GEMM: C = act( A0@Wt0^T + A1@Wt1^T + bias ), all operands bf16 ----
// N fixed at 256. Block = 4 waves = one 16-row stripe; each wave does 16x64 (4 tiles).
__global__ __launch_bounds__(128)
void sage_gemm_kernel(const unsigned short* __restrict__ A0,
                      const unsigned short* __restrict__ A1,
                      const unsigned short* __restrict__ Wt0,
                      const unsigned short* __restrict__ Wt1,
                      const float* __restrict__ bias, float* __restrict__ C,
                      int M, int K, int relu) {
  const int N = 256;
  int wave = threadIdx.x >> 5;               // 0..3 -> nTiles 4w..4w+3
  int lane = threadIdx.x & 31;
  int mT = blockIdx.x;
  int half = lane >> 4, l16 = lane & 15;
  int row  = mT * 16 + l16;
  int koff = half * 8;

  v8f acc[4];
  #pragma unroll
  for (int t = 0; t < 4; ++t)
    acc[t] = (v8f){0.f, 0.f, 0.f, 0.f, 0.f, 0.f, 0.f, 0.f};

  const unsigned short* a0 = A0 + (size_t)row * K;
  const unsigned short* a1 = A1 + (size_t)row * K;
  int colBase = wave * 64 + l16;

  #pragma unroll
  for (int s = 0; s < 2; ++s) {
    const unsigned short* arow = s ? a1 : a0;
    const unsigned short* Wt   = s ? Wt1 : Wt0;
    const unsigned short* b0 = Wt + (size_t)(colBase +  0) * K;
    const unsigned short* b1 = Wt + (size_t)(colBase + 16) * K;
    const unsigned short* b2 = Wt + (size_t)(colBase + 32) * K;
    const unsigned short* b3 = Wt + (size_t)(colBase + 48) * K;
    for (int kb = 0; kb < K; kb += 32) {
      v16bf av = load_frag(arow, kb, koff);
      acc[0] = __builtin_amdgcn_wmma_f32_16x16x32_bf16(
          false, av, false, load_frag(b0, kb, koff), (short)0, acc[0], false, false);
      acc[1] = __builtin_amdgcn_wmma_f32_16x16x32_bf16(
          false, av, false, load_frag(b1, kb, koff), (short)0, acc[1], false, false);
      acc[2] = __builtin_amdgcn_wmma_f32_16x16x32_bf16(
          false, av, false, load_frag(b2, kb, koff), (short)0, acc[2], false, false);
      acc[3] = __builtin_amdgcn_wmma_f32_16x16x32_bf16(
          false, av, false, load_frag(b3, kb, koff), (short)0, acc[3], false, false);
    }
  }

  #pragma unroll
  for (int t = 0; t < 4; ++t) {
    int col = colBase + t * 16;
    float bv = bias[col];
    float* cp = C + (size_t)(mT * 16 + 8 * half) * N + col;
    #pragma unroll
    for (int r = 0; r < 8; ++r) {
      float v = acc[t][r] + bv;
      if (relu) v = fmaxf(v, 0.f);
      cp[(size_t)r * N] = v;
    }
  }
}

extern "C" void kernel_launch(void* const* d_in, const int* in_sizes, int n_in,
                              void* d_out, int out_size, void* d_ws, size_t ws_size,
                              hipStream_t stream) {
  const float* x_user = (const float*)d_in[0];
  const float* x_item = (const float*)d_in[1];
  const float* W1l_ui = (const float*)d_in[2];
  const float* W1r_ui = (const float*)d_in[3];
  const float* b1_ui  = (const float*)d_in[4];
  const float* W1l_iu = (const float*)d_in[5];
  const float* W1r_iu = (const float*)d_in[6];
  const float* b1_iu  = (const float*)d_in[7];
  const float* W2l_ui = (const float*)d_in[8];
  const float* W2r_ui = (const float*)d_in[9];
  const float* b2_ui  = (const float*)d_in[10];
  const float* W2l_iu = (const float*)d_in[11];
  const float* W2r_iu = (const float*)d_in[12];
  const float* b2_iu  = (const float*)d_in[13];
  const int* src_ui = (const int*)d_in[14];
  const int* dst_ui = (const int*)d_in[15];
  const int* src_iu = (const int*)d_in[16];
  const int* dst_iu = (const int*)d_in[17];

  // ---- workspace layout ----
  char* base = (char*)d_ws;
  size_t off = 0;
  auto alloc = [&](size_t bytes) { size_t o = off; off = (off + bytes + 255) & ~(size_t)255; return o; };

  float* rinv_user = (float*)(base + alloc((size_t)N_USERN * 4));
  float* rinv_item = (float*)(base + alloc((size_t)N_ITEMN * 4));
  unsigned short* wt1l_ui = (unsigned short*)(base + alloc((size_t)D_INN * HN * 2));
  unsigned short* wt1r_ui = (unsigned short*)(base + alloc((size_t)D_INN * HN * 2));
  unsigned short* wt1l_iu = (unsigned short*)(base + alloc((size_t)D_INN * HN * 2));
  unsigned short* wt1r_iu = (unsigned short*)(base + alloc((size_t)D_INN * HN * 2));
  unsigned short* wt2l_ui = (unsigned short*)(base + alloc((size_t)HN * HN * 2));
  unsigned short* wt2r_ui = (unsigned short*)(base + alloc((size_t)HN * HN * 2));
  unsigned short* wt2l_iu = (unsigned short*)(base + alloc((size_t)HN * HN * 2));
  unsigned short* wt2r_iu = (unsigned short*)(base + alloc((size_t)HN * HN * 2));
  float* agg_u  = (float*)(base + alloc((size_t)N_USERN * HN * 4));  // layer1 K=128 / layer2 K=256
  float* agg_i  = (float*)(base + alloc((size_t)N_ITEMN * HN * 4));
  float* h_user = (float*)(base + alloc((size_t)N_USERN * HN * 4));
  float* h_item = (float*)(base + alloc((size_t)N_ITEMN * HN * 4));
  unsigned short* a0s = (unsigned short*)(base + alloc((size_t)N_USERN * HN * 2)); // bf16 staging
  unsigned short* a1s = (unsigned short*)(base + alloc((size_t)N_USERN * HN * 2));

  float* o_user = (float*)d_out;
  float* o_item = (float*)d_out + (size_t)N_USERN * HN;

  const int TB = 256;
  auto cvtGrid = [&](long long m, long long k) { return (int)((m * k / 4 + TB - 1) / TB); };

  // ---- degree counts -> reciprocals ----
  hipMemsetAsync(rinv_user, 0, (size_t)N_USERN * 4, stream);
  hipMemsetAsync(rinv_item, 0, (size_t)N_ITEMN * 4, stream);
  count_kernel<<<(NEDGE + TB - 1) / TB, TB, 0, stream>>>(dst_ui, rinv_item, NEDGE);
  count_kernel<<<(NEDGE + TB - 1) / TB, TB, 0, stream>>>(dst_iu, rinv_user, NEDGE);
  rinv_kernel<<<(N_ITEMN + TB - 1) / TB, TB, 0, stream>>>(rinv_item, N_ITEMN);
  rinv_kernel<<<(N_USERN + TB - 1) / TB, TB, 0, stream>>>(rinv_user, N_USERN);

  // ---- weights: fp32 [KxN] -> bf16 [NxK] ----
  int e1 = D_INN * HN, e2 = HN * HN;
  wconv_kernel<<<(e1 + TB - 1) / TB, TB, 0, stream>>>(W1l_ui, wt1l_ui, D_INN, HN);
  wconv_kernel<<<(e1 + TB - 1) / TB, TB, 0, stream>>>(W1r_ui, wt1r_ui, D_INN, HN);
  wconv_kernel<<<(e1 + TB - 1) / TB, TB, 0, stream>>>(W1l_iu, wt1l_iu, D_INN, HN);
  wconv_kernel<<<(e1 + TB - 1) / TB, TB, 0, stream>>>(W1r_iu, wt1r_iu, D_INN, HN);
  wconv_kernel<<<(e2 + TB - 1) / TB, TB, 0, stream>>>(W2l_ui, wt2l_ui, HN, HN);
  wconv_kernel<<<(e2 + TB - 1) / TB, TB, 0, stream>>>(W2r_ui, wt2r_ui, HN, HN);
  wconv_kernel<<<(e2 + TB - 1) / TB, TB, 0, stream>>>(W2l_iu, wt2l_iu, HN, HN);
  wconv_kernel<<<(e2 + TB - 1) / TB, TB, 0, stream>>>(W2r_iu, wt2r_iu, HN, HN);

  // ---- layer 1: scatter-sum (D=128) ----
  hipMemsetAsync(agg_i, 0, (size_t)N_ITEMN * D_INN * 4, stream);
  hipMemsetAsync(agg_u, 0, (size_t)N_USERN * D_INN * 4, stream);
  {
    long long n1 = (long long)NEDGE * (D_INN / 4);
    int g = (int)((n1 + TB - 1) / TB);
    scatter_kernel<<<g, TB, 0, stream>>>(x_user, src_ui, dst_ui, agg_i, NEDGE, D_INN);
    scatter_kernel<<<g, TB, 0, stream>>>(x_item, src_iu, dst_iu, agg_u, NEDGE, D_INN);
  }

  // ---- layer 1 GEMMs: stage bf16 (mean-scale folded in), then WMMA ----
  cvt_kernel<<<cvtGrid(N_ITEMN, D_INN), TB, 0, stream>>>(agg_i, rinv_item, a0s, N_ITEMN, D_INN);
  cvt_kernel<<<cvtGrid(N_ITEMN, D_INN), TB, 0, stream>>>(x_item, nullptr,  a1s, N_ITEMN, D_INN);
  sage_gemm_kernel<<<N_ITEMN / 16, 128, 0, stream>>>(
      a0s, a1s, wt1l_ui, wt1r_ui, b1_ui, h_item, N_ITEMN, D_INN, 1);

  cvt_kernel<<<cvtGrid(N_USERN, D_INN), TB, 0, stream>>>(agg_u, rinv_user, a0s, N_USERN, D_INN);
  cvt_kernel<<<cvtGrid(N_USERN, D_INN), TB, 0, stream>>>(x_user, nullptr,  a1s, N_USERN, D_INN);
  sage_gemm_kernel<<<N_USERN / 16, 128, 0, stream>>>(
      a0s, a1s, wt1l_iu, wt1r_iu, b1_iu, h_user, N_USERN, D_INN, 1);

  // ---- layer 2: scatter-sum (D=256) ----
  hipMemsetAsync(agg_i, 0, (size_t)N_ITEMN * HN * 4, stream);
  hipMemsetAsync(agg_u, 0, (size_t)N_USERN * HN * 4, stream);
  {
    long long n2 = (long long)NEDGE * (HN / 4);
    int g = (int)((n2 + TB - 1) / TB);
    scatter_kernel<<<g, TB, 0, stream>>>(h_user, src_ui, dst_ui, agg_i, NEDGE, HN);
    scatter_kernel<<<g, TB, 0, stream>>>(h_item, src_iu, dst_iu, agg_u, NEDGE, HN);
  }

  // ---- layer 2 GEMMs -> d_out ----
  cvt_kernel<<<cvtGrid(N_ITEMN, HN), TB, 0, stream>>>(agg_i, rinv_item, a0s, N_ITEMN, HN);
  cvt_kernel<<<cvtGrid(N_ITEMN, HN), TB, 0, stream>>>(h_item, nullptr,  a1s, N_ITEMN, HN);
  sage_gemm_kernel<<<N_ITEMN / 16, 128, 0, stream>>>(
      a0s, a1s, wt2l_ui, wt2r_ui, b2_ui, o_item, N_ITEMN, HN, 0);

  cvt_kernel<<<cvtGrid(N_USERN, HN), TB, 0, stream>>>(agg_u, rinv_user, a0s, N_USERN, HN);
  cvt_kernel<<<cvtGrid(N_USERN, HN), TB, 0, stream>>>(h_user, nullptr,  a1s, N_USERN, HN);
  sage_gemm_kernel<<<N_USERN / 16, 128, 0, stream>>>(
      a0s, a1s, wt2l_iu, wt2r_iu, b2_iu, o_user, N_USERN, HN, 0);
}
